// RegionProposalNetwork_66795331387624
// MI455X (gfx1250) — matrix operand order repaired and checked by reading, not compiled, not run on hardware
//
#include <hip/hip_runtime.h>
#include <hip/hip_bf16.h>

typedef __attribute__((ext_vector_type(16))) _Float16 v16h;
typedef __attribute__((ext_vector_type(8)))  float    v8f;

union FragU { unsigned int u[8]; uint4 q[2]; v16h h; };

// ---------------- workspace layout (bytes) ----------------
#define OFF_XP    0u         // 34*34*256 f16 padded activations [cell][ci] (591872)
#define OFF_W1    591872u    // 512*2304 f16 weights [co][kpos*256+ci]      (2359296)
#define OFF_WC    2951168u   // 48*512 f16 combined head weights [o][c]     (49152)
#define OFF_H     3000320u   // 1024*512 f16 hidden [pix][co]               (1048576)
#define OFF_RAW   4048896u   // 1024*48 f32 head outputs                    (196608)
#define OFF_BOX   4245504u   // 9216 float4 boxes                           (147456)
#define OFF_SC    4392960u   // 9216 f32 scores                             (36864)
#define OFF_VAL   4429824u   // 9216 int valid                              (36864)
#define OFF_KEYS  4466688u   // 16384 u64 sort keys                         (131072)
#define OFF_ORD   4597760u   // 9216 int sorted order                       (65536)
#define OFF_KEEP  4663296u   // 9216 int keep mask                          (36864)

#define NBOX 9216
#define NPAD 16384

// ---------------- packing kernels ----------------
// Padded activations: cell = (y+1)*34 + (x+1), halo cells are zero.
__global__ void pack_x(const float* __restrict__ fmap, _Float16* __restrict__ XP) {
    int idx = blockIdx.x * 256 + threadIdx.x;            // 295936 = 34*34*256
    if (idx >= 295936) return;
    int ci   = idx & 255;
    int cell = idx >> 8;
    int yp = cell / 34, xp = cell - yp * 34;
    int y = yp - 1, x = xp - 1;
    float v = 0.f;
    if ((unsigned)y < 32u && (unsigned)x < 32u)
        v = fmap[ci * 1024 + (y << 5) + x];
    XP[idx] = (_Float16)v;
}

__global__ void pack_w1(const float* __restrict__ w1, _Float16* __restrict__ W1h) {
    int idx = blockIdx.x * 256 + threadIdx.x;            // 1179648
    if (idx >= 1179648) return;
    int co = idx / 2304;
    int r  = idx - co * 2304;
    int ci = r / 9;
    int kp = r - ci * 9;
    W1h[co * 2304 + kp * 256 + ci] = (_Float16)w1[idx];
}

__global__ void pack_wc(const float* __restrict__ w2, const float* __restrict__ w3,
                        _Float16* __restrict__ WCh) {
    int idx = blockIdx.x * 256 + threadIdx.x;            // 24576
    if (idx >= 24576) return;
    int o = idx >> 9;
    int c = idx & 511;
    float v = 0.f;
    if (o < 9)       v = w2[o * 512 + c];
    else if (o < 45) v = w3[(o - 9) * 512 + c];
    WCh[idx] = (_Float16)v;
}

// ---------------- conv1: implicit GEMM, 16x16x32 f16 WMMA, 16x64 tile per wave -------
// M = 1024 pixels, N = 512 cout, K = 2304 (= 9 taps * 256 ci).
// Ping-pong software pipeline: loads for chunk t+1 issue before WMMAs of chunk t.
__device__ __forceinline__ void conv1_load_chunk(const _Float16* __restrict__ X,
                                                 const _Float16* __restrict__ W,
                                                 int t, int py, int px, int halfA,
                                                 size_t wBase, FragU& a, FragU (&b)[4]) {
    const int kpos = t >> 3;               // 3x3 tap index
    const int ciB  = (t & 7) << 5;         // ci chunk base
    const int dy = kpos / 3, dx = kpos % 3;            // 0..2 (halo-shifted)
    const int cell = (py + dy) * 34 + (px + dx);       // padded cell
    const _Float16* xp  = X + cell * 256 + ciB + halfA;
    const _Float16* wp0 = W + wBase + kpos * 256 + ciB;

    __builtin_prefetch(wp0 + 32, 0, 3);    // next ci chunk -> global_prefetch_b8

    a.q[0] = *(const uint4*)(xp);          // K 0..7  (this half)
    a.q[1] = *(const uint4*)(xp + 16);     // K 16..23 (this half)
#pragma unroll
    for (int s = 0; s < 4; ++s) {
        const _Float16* wp = wp0 + s * (16 * 2304);
        b[s].q[0] = *(const uint4*)(wp);
        b[s].q[1] = *(const uint4*)(wp + 16);
    }
}

__global__ void __launch_bounds__(256) conv1_wmma(const _Float16* __restrict__ X,
                                                  const _Float16* __restrict__ W,
                                                  const float* __restrict__ bias,
                                                  _Float16* __restrict__ H) {
    const int lane  = threadIdx.x & 31;
    const int wave  = threadIdx.x >> 5;
    const int tile  = blockIdx.x * 8 + wave;   // 512 wave-tiles (64 m x 8 n-groups)
    const int mTile = tile >> 3;               // 0..63
    const int nGrp  = tile & 7;                // 0..7 (4 cout tiles each)

    const int halfA = (lane < 16) ? 0 : 8;     // 16-bit A 16x32 ISA layout
    const int halfB = (lane < 16) ? 0 : 8;     // B 32x16 layout
    const int nn    = lane & 15;
    const int co0   = nGrp * 64 + nn;          // column of sub-tile 0
    const size_t wBase = (size_t)co0 * 2304 + halfB;
    const int mrow  = lane & 15;
    const int p     = mTile * 16 + mrow;
    const int py    = p >> 5, px = p & 31;

    v8f acc[4] = {{}, {}, {}, {}};
    FragU Pa, Pb[4], Qa, Qb[4];
    conv1_load_chunk(X, W, 0, py, px, halfA, wBase, Pa, Pb);

    int t = 0;
    for (; t + 2 < 72; t += 2) {               // pairs (t, t+1); loads lead WMMAs by 1 chunk
        conv1_load_chunk(X, W, t + 1, py, px, halfA, wBase, Qa, Qb);
#pragma unroll
        for (int s = 0; s < 4; ++s)
            acc[s] = __builtin_amdgcn_wmma_f32_16x16x32_f16(false, Pa.h, false, Pb[s].h,
                                                            (short)0, acc[s], false, false);
        conv1_load_chunk(X, W, t + 2, py, px, halfA, wBase, Pa, Pb);
#pragma unroll
        for (int s = 0; s < 4; ++s)
            acc[s] = __builtin_amdgcn_wmma_f32_16x16x32_f16(false, Qa.h, false, Qb[s].h,
                                                            (short)0, acc[s], false, false);
    }
    conv1_load_chunk(X, W, 71, py, px, halfA, wBase, Qa, Qb);
#pragma unroll
    for (int s = 0; s < 4; ++s)                // chunk 70
        acc[s] = __builtin_amdgcn_wmma_f32_16x16x32_f16(false, Pa.h, false, Pb[s].h,
                                                        (short)0, acc[s], false, false);
#pragma unroll
    for (int s = 0; s < 4; ++s)                // chunk 71
        acc[s] = __builtin_amdgcn_wmma_f32_16x16x32_f16(false, Qa.h, false, Qb[s].h,
                                                        (short)0, acc[s], false, false);

    const int rbase = (lane < 16) ? 0 : 8;
#pragma unroll
    for (int s = 0; s < 4; ++s) {
        const int co = co0 + s * 16;
        const float bv = bias[co];
#pragma unroll
        for (int v = 0; v < 8; ++v) {
            int pp = mTile * 16 + v + rbase;
            float val = acc[s][v] + bv;
            val = val > 0.f ? val : 0.f;       // ReLU
            H[pp * 512 + co] = (_Float16)val;
        }
    }
}

// ---------------- heads: M=1024, N=48 (9 score + 36 offset + 3 pad), K=512 -----------
// One wave handles all 3 cout tiles of one m-tile; same ping-pong pipeline.
__device__ __forceinline__ void gemm2_load_chunk(const _Float16* __restrict__ H,
                                                 const _Float16* __restrict__ WC,
                                                 int t, int p, int halfA, int halfB, int nn,
                                                 FragU& a, FragU (&b)[3]) {
    const int kB = t << 5;
    const _Float16* hp = H + p * 512 + kB + halfA;
    a.q[0] = *(const uint4*)(hp);
    a.q[1] = *(const uint4*)(hp + 16);
#pragma unroll
    for (int s = 0; s < 3; ++s) {
        const _Float16* wp = WC + (nn + s * 16) * 512 + kB + halfB;
        b[s].q[0] = *(const uint4*)(wp);
        b[s].q[1] = *(const uint4*)(wp + 16);
    }
}

__global__ void __launch_bounds__(256) gemm2_wmma(const _Float16* __restrict__ H,
                                                  const _Float16* __restrict__ WC,
                                                  float* __restrict__ RAW) {
    const int lane  = threadIdx.x & 31;
    const int wave  = threadIdx.x >> 5;
    const int mTile = blockIdx.x * 8 + wave;   // 0..63

    const int halfA = (lane < 16) ? 0 : 8;
    const int halfB = (lane < 16) ? 0 : 8;
    const int nn    = lane & 15;
    const int mrow  = lane & 15;
    const int p     = mTile * 16 + mrow;

    v8f acc[3] = {{}, {}, {}};
    FragU Pa, Pb[3], Qa, Qb[3];
    gemm2_load_chunk(H, WC, 0, p, halfA, halfB, nn, Pa, Pb);

    int t = 0;
    for (; t + 2 < 16; t += 2) {
        gemm2_load_chunk(H, WC, t + 1, p, halfA, halfB, nn, Qa, Qb);
#pragma unroll
        for (int s = 0; s < 3; ++s)
            acc[s] = __builtin_amdgcn_wmma_f32_16x16x32_f16(false, Pa.h, false, Pb[s].h,
                                                            (short)0, acc[s], false, false);
        gemm2_load_chunk(H, WC, t + 2, p, halfA, halfB, nn, Pa, Pb);
#pragma unroll
        for (int s = 0; s < 3; ++s)
            acc[s] = __builtin_amdgcn_wmma_f32_16x16x32_f16(false, Qa.h, false, Qb[s].h,
                                                            (short)0, acc[s], false, false);
    }
    gemm2_load_chunk(H, WC, 15, p, halfA, halfB, nn, Qa, Qb);
#pragma unroll
    for (int s = 0; s < 3; ++s)                // chunk 14
        acc[s] = __builtin_amdgcn_wmma_f32_16x16x32_f16(false, Pa.h, false, Pb[s].h,
                                                        (short)0, acc[s], false, false);
#pragma unroll
    for (int s = 0; s < 3; ++s)                // chunk 15
        acc[s] = __builtin_amdgcn_wmma_f32_16x16x32_f16(false, Qa.h, false, Qb[s].h,
                                                        (short)0, acc[s], false, false);

    const int rbase = (lane < 16) ? 0 : 8;
#pragma unroll
    for (int s = 0; s < 3; ++s) {
#pragma unroll
        for (int v = 0; v < 8; ++v) {
            int pp = mTile * 16 + v + rbase;
            RAW[pp * 48 + nn + s * 16] = acc[s][v];
        }
    }
}

// ---------------- box decode + sort-key build ----------------
__global__ void decode_kernel(const float* __restrict__ RAW,
                              const float* __restrict__ b2, const float* __restrict__ b3,
                              float4* __restrict__ BOX, float* __restrict__ SC,
                              int* __restrict__ VALID, unsigned long long* __restrict__ KEYS) {
    int i = blockIdx.x * 256 + threadIdx.x;    // grid covers NPAD
    if (i >= NPAD) return;
    if (i >= NBOX) { KEYS[i] = ~0ull; return; }

    int p = i / 9, a = i - p * 9;
    int y = p >> 5, x = p & 31;
    int s = a / 3, r = a - s * 3;
    const float sizes[3] = {32.f, 64.f, 128.f};
    const float sqr[3]   = {0.70710678f, 1.f, 1.41421356f};
    float aw = sizes[s] / sqr[r];
    float ah = sizes[s] * sqr[r];
    float acx = (x + 0.5f) * 16.f;
    float acy = (y + 0.5f) * 16.f;

    float logit = RAW[p * 48 + a] + b2[a];
    float score = 1.f / (1.f + __expf(-logit));
    float tx = RAW[p * 48 + 9 + a * 4 + 0] + b3[a * 4 + 0];
    float ty = RAW[p * 48 + 9 + a * 4 + 1] + b3[a * 4 + 1];
    float tw = RAW[p * 48 + 9 + a * 4 + 2] + b3[a * 4 + 2];
    float th = RAW[p * 48 + 9 + a * 4 + 3] + b3[a * 4 + 3];

    float cx = acx + tx * aw, cy = acy + ty * ah;
    float bw = aw * __expf(tw), bh = ah * __expf(th);
    float x1 = fminf(fmaxf(cx - 0.5f * bw, 0.f), 512.f);
    float y1 = fminf(fmaxf(cy - 0.5f * bh, 0.f), 512.f);
    float x2 = fminf(fmaxf(cx + 0.5f * bw, 0.f), 512.f);
    float y2 = fminf(fmaxf(cy + 0.5f * bh, 0.f), 512.f);
    int valid = (x2 - x1 >= 0.001f) && (y2 - y1 >= 0.001f) && (score >= 0.5f);

    BOX[i] = make_float4(x1, y1, x2, y2);
    SC[i] = score;
    VALID[i] = valid;

    float skey = valid ? score : -__builtin_inff();
    unsigned int bits = __float_as_uint(skey);
    unsigned int u = (bits & 0x80000000u) ? ~bits : (bits | 0x80000000u); // ascending order
    KEYS[i] = ((unsigned long long)(~u) << 32) | (unsigned int)i;         // descending score, idx tiebreak
}

// ---------------- single-block bitonic sort in 128 KB LDS ----------------
__global__ void __launch_bounds__(1024) sort_kernel(const unsigned long long* __restrict__ KEYS,
                                                    int* __restrict__ ORDER) {
    extern __shared__ unsigned long long sk[];
    for (int t = threadIdx.x; t < NPAD; t += 1024) sk[t] = KEYS[t];
    __syncthreads();
    for (int k = 2; k <= NPAD; k <<= 1) {
        for (int j = k >> 1; j > 0; j >>= 1) {
            for (int t = threadIdx.x; t < NPAD; t += 1024) {
                int ixj = t ^ j;
                if (ixj > t) {
                    bool up = ((t & k) == 0);
                    unsigned long long A = sk[t], B = sk[ixj];
                    if ((A > B) == up) { sk[t] = B; sk[ixj] = A; }
                }
            }
            __syncthreads();
        }
    }
    for (int t = threadIdx.x; t < NBOX; t += 1024)
        ORDER[t] = (int)(sk[t] & 0xFFFFFFFFull);
}

// ---------------- greedy NMS: all boxes resident in ~166 KB LDS (CDNA5 320KB/WGP) ---------
__global__ void __launch_bounds__(1024) nms_kernel(const int* __restrict__ ORDER,
                                                   const float4* __restrict__ BOX,
                                                   const int* __restrict__ VALID,
                                                   int* __restrict__ KEEP) {
    extern __shared__ unsigned char smem[];
    float4* sb         = (float4*)smem;               // 9216 * 16 B
    unsigned char* vf  = smem + NBOX * 16;            // 9216 B
    unsigned char* sup = vf + NBOX;                   // 9216 B

    for (int t = threadIdx.x; t < NBOX; t += 1024) {
        int o = ORDER[t];
        sb[t]  = BOX[o];
        vf[t]  = (unsigned char)VALID[o];
        sup[t] = 0;
    }
    __syncthreads();

    for (int i = 0; i < NBOX; ++i) {
        bool kept = (!sup[i]) && vf[i];
        if (kept) {
            float4 bi = sb[i];
            float areai = (bi.z - bi.x) * (bi.w - bi.y);
            for (int j = i + 1 + (int)threadIdx.x; j < NBOX; j += 1024) {
                float4 bj = sb[j];
                float ix1 = fmaxf(bi.x, bj.x), iy1 = fmaxf(bi.y, bj.y);
                float ix2 = fminf(bi.z, bj.z), iy2 = fminf(bi.w, bj.w);
                float inter = fmaxf(ix2 - ix1, 0.f) * fmaxf(iy2 - iy1, 0.f);
                float areaj = (bj.z - bj.x) * (bj.w - bj.y);
                float iou = inter / fmaxf(areai + areaj - inter, 1e-9f);
                if (iou > 0.7f) sup[j] = 1;
            }
        }
        __syncthreads();
    }

    for (int t = threadIdx.x; t < NBOX; t += 1024) {
        int o = ORDER[t];
        KEEP[o] = ((!sup[t]) && vf[t]) ? 1 : 0;
    }
}

// ---------------- final gather ----------------
__global__ void output_kernel(const float4* __restrict__ BOX, const float* __restrict__ SC,
                              const int* __restrict__ KEEP, float* __restrict__ out) {
    int i = blockIdx.x * 256 + threadIdx.x;
    if (i >= NBOX) return;
    float k = KEEP[i] ? 1.f : 0.f;
    float4 b = BOX[i];
    out[i * 5 + 0] = b.x * k;
    out[i * 5 + 1] = b.y * k;
    out[i * 5 + 2] = b.z * k;
    out[i * 5 + 3] = b.w * k;
    out[i * 5 + 4] = SC[i] * k;
}

extern "C" void kernel_launch(void* const* d_in, const int* in_sizes, int n_in,
                              void* d_out, int out_size, void* d_ws, size_t ws_size,
                              hipStream_t stream) {
    const float* fmap = (const float*)d_in[0];
    const float* w1   = (const float*)d_in[1];
    const float* b1   = (const float*)d_in[2];
    const float* w2   = (const float*)d_in[3];
    const float* b2   = (const float*)d_in[4];
    const float* w3   = (const float*)d_in[5];
    const float* b3   = (const float*)d_in[6];
    (void)in_sizes; (void)n_in; (void)out_size; (void)ws_size;

    char* ws = (char*)d_ws;
    _Float16* XP  = (_Float16*)(ws + OFF_XP);
    _Float16* W1h = (_Float16*)(ws + OFF_W1);
    _Float16* WCh = (_Float16*)(ws + OFF_WC);
    _Float16* Hh  = (_Float16*)(ws + OFF_H);
    float*    RAW = (float*)(ws + OFF_RAW);
    float4*   BOX = (float4*)(ws + OFF_BOX);
    float*    SC  = (float*)(ws + OFF_SC);
    int*      VAL = (int*)(ws + OFF_VAL);
    unsigned long long* KEYS = (unsigned long long*)(ws + OFF_KEYS);
    int*      ORD = (int*)(ws + OFF_ORD);
    int*      KEEP = (int*)(ws + OFF_KEEP);
    float*    out = (float*)d_out;

    pack_x <<<1156, 256, 0, stream>>>(fmap, XP);
    pack_w1<<<4608, 256, 0, stream>>>(w1, W1h);
    pack_wc<<<96,   256, 0, stream>>>(w2, w3, WCh);

    conv1_wmma<<<64, 256, 0, stream>>>(XP, W1h, b1, Hh);
    gemm2_wmma<<<8,  256, 0, stream>>>(Hh, WCh, RAW);

    decode_kernel<<<64, 256, 0, stream>>>(RAW, b2, b3, BOX, SC, VAL, KEYS);
    sort_kernel<<<1, 1024, NPAD * sizeof(unsigned long long), stream>>>(KEYS, ORD);
    nms_kernel<<<1, 1024, NBOX * 16 + 2 * NBOX, stream>>>(ORD, BOX, VAL, KEEP);
    output_kernel<<<36, 256, 0, stream>>>(BOX, SC, KEEP, out);
}
